// SegmentationAttentionSeparateModule_67843303408119
// MI455X (gfx1250) — matrix operand order repaired and checked by reading, not compiled, not run on hardware
//
#include <hip/hip_runtime.h>

typedef __attribute__((ext_vector_type(16))) __bf16 v16bf;
typedef __attribute__((ext_vector_type(8)))  float  v8f;
typedef __attribute__((ext_vector_type(4)))  unsigned int v4u;
typedef __attribute__((ext_vector_type(8)))  int v8i32;
typedef __attribute__((ext_vector_type(4)))  int v4i32;

#define BB 4
#define DKD 64
#define DVD 512
#define NQ 4096
#define MKM 8192
#define BN 64          // queries per workgroup (4 n-tiles of 16)
#define BM 32          // keys per chunk (2 m-subtiles of 16)
#define NCH (MKM / BM) // 256 chunks
#define KST 66         // LDS row stride (elems) for [x][DKD] tiles = 128B + 4B pad
#define VST 34         // LDS row stride (elems) for [x][BM] tiles  =  64B + 4B pad
#define NEG_BIG (-1.0e30f)

union FragU { unsigned int u[8]; v16bf b; };

__device__ __forceinline__ unsigned short f2bf(float f) {
  unsigned int u = __float_as_uint(f);
  u += 0x7FFFu + ((u >> 16) & 1u);   // round-to-nearest-even
  return (unsigned short)(u >> 16);
}

// ---------------- TDM: 2D bf16 tile -> LDS with row padding ----------------
// pad_interval code: 3 -> pad after 16 DWORDs (64B rows), 4 -> after 32 DWORDs (128B rows)
// pad_amount fixed at code 0 (1 DWORD = 4B) -> LDS row stride = row + 4B.
__device__ __forceinline__ void tdm_load_2d(unsigned lds_off, const void* gptr,
                                            unsigned dim0, unsigned dim1,
                                            unsigned tile0, unsigned tile1,
                                            unsigned stride0, unsigned pad_code) {
  unsigned long long ga = (unsigned long long)gptr;
  v4u g0;
  g0[0] = 1u;                                            // count=1 (valid user D#)
  g0[1] = lds_off;                                       // lds_addr (bytes)
  g0[2] = (unsigned)(ga & 0xFFFFFFFFu);                  // global_addr[31:0]
  g0[3] = (unsigned)((ga >> 32) & 0x01FFFFFFu) | (2u << 30); // addr[56:32] | type=2
  v8i32 g1;
  g1[0] = (int)((1u << 16) | (1u << 20) | (pad_code << 22)); // data_size=2B, pad_en, pad_amt=1DW
  g1[1] = (int)((dim0 & 0xFFFFu) << 16);                 // tensor_dim0[15:0]
  g1[2] = (int)((dim0 >> 16) | ((dim1 & 0xFFFFu) << 16));// dim0[31:16] | dim1[15:0]
  g1[3] = (int)((dim1 >> 16) | (tile0 << 16));           // dim1[31:16] | tile_dim0
  g1[4] = (int)tile1;                                    // tile_dim1 (tile_dim2=0)
  g1[5] = (int)stride0;                                  // tensor_dim0_stride[31:0]
  g1[6] = 0;                                             // stride0 hi | stride1 lo
  g1[7] = 0;
  v4i32 z4; z4[0] = z4[1] = z4[2] = z4[3] = 0;
  v8i32 z8;
#pragma unroll
  for (int e = 0; e < 8; ++e) z8[e] = 0;
  // 6-arg variant (clang-23 / therock HIP headers)
  __builtin_amdgcn_tensor_load_to_lds(g0, g1, z4, z4, z8, 0);
}

// ---------------- pre-conversion kernels (f32 -> bf16 once) ----------------
__global__ __launch_bounds__(256)
void conv_q_kernel(const float* __restrict__ qk, unsigned short* __restrict__ qh) {
  // qk: [B][DK][NQ] f32  ->  qh: [B][NQ][DK] bf16, pre-scaled by 1/sqrt(DK)
  long i = (long)blockIdx.x * 256 + threadIdx.x;
  if (i >= (long)BB * DKD * NQ) return;
  int n = (int)(i % NQ);
  int d = (int)((i / NQ) % DKD);
  int b = (int)(i / ((long)NQ * DKD));
  qh[((long)b * NQ + n) * DKD + d] = f2bf(qk[i] * 0.125f);
}

__global__ __launch_bounds__(256)
void conv_k_kernel(const float* __restrict__ mk, unsigned short* __restrict__ kh) {
  // mk: [B][DK][MK] f32  ->  kh: [B][MK][DK] bf16 (transposed for WMMA A-fragments)
  long i = (long)blockIdx.x * 256 + threadIdx.x;
  if (i >= (long)BB * DKD * MKM) return;
  int m = (int)(i % MKM);
  int d = (int)((i / MKM) % DKD);
  int b = (int)(i / ((long)MKM * DKD));
  kh[((long)b * MKM + m) * DKD + d] = f2bf(mk[i]);
}

__global__ __launch_bounds__(256)
void conv_v_kernel(const float* __restrict__ mv, unsigned int* __restrict__ vh) {
  // mv: [B][DV][MK] f32 -> vh: same layout bf16 (packed pairs)
  const long npairs = (long)BB * DVD * MKM / 2;
  for (long i = (long)blockIdx.x * 256 + threadIdx.x; i < npairs; i += (long)gridDim.x * 256) {
    float a = mv[2 * i], b = mv[2 * i + 1];
    vh[i] = (unsigned int)f2bf(a) | ((unsigned int)f2bf(b) << 16);
  }
}

// ---------------- WMMA fragment gathers from padded LDS ----------------
__device__ __forceinline__ v16bf load_frag_a(const unsigned short* lds, int row_base,
                                             int st, int koff, int lane) {
  FragU f;
  const unsigned short* p = lds + (row_base + (lane & 15)) * st + koff + ((lane >> 4) * 16);
#pragma unroll
  for (int r = 0; r < 8; ++r) {
    int c = 2 * r;
    f.u[r] = *(const unsigned int*)(p + c);
  }
  return f.b;
}

__device__ __forceinline__ v16bf load_frag_b(const unsigned short* lds, int col_base,
                                             int st, int koff, int lane) {
  FragU f;
  const unsigned short* p = lds + (col_base + (lane & 15)) * st + koff + ((lane >> 4) * 16);
#pragma unroll
  for (int r = 0; r < 8; ++r)
    f.u[r] = *(const unsigned int*)(p + 2 * r);
  return f.b;
}

// ---------------- main flash-attention kernel (double-buffered TDM) ----------------
__global__ __launch_bounds__(256)
void maskread_attn_kernel(const unsigned short* __restrict__ qh,  // [B][NQ][DK] bf16
                          const unsigned char* __restrict__ qm,   // [B][NQ]
                          const unsigned short* __restrict__ kh,  // [B][MK][DK] bf16
                          const unsigned short* __restrict__ vh,  // [B][DV][MK] bf16
                          const unsigned char* __restrict__ mmk,  // [B][MK]
                          float* __restrict__ out)                // [B][DV][NQ] f32
{
  __shared__ unsigned short Vl[2][DVD * VST]; // V chunk [v][m], TDM-padded rows
  __shared__ unsigned short Kl[2][BM * KST];  // K chunk [m][d]
  __shared__ unsigned short Ql[BN * KST];     // Q tile  [n][d]
  __shared__ unsigned short Pl[BN * VST];     // P       [n][m]
  __shared__ float biasS[2][BM];
  __shared__ float mrun[BN], lrun[BN], alphaS[BN], qscS[BN];
  __shared__ float pmaxS[2][BN], psumS[2][BN];

  const int tid  = threadIdx.x;
  const int wave = tid >> 5;
  const int lane = tid & 31;
  const int b    = blockIdx.x >> 6;           // NQ/BN = 64 tiles per batch
  const int n0   = (blockIdx.x & 63) * BN;
  const int kh2  = lane >> 4;

  const unsigned short* kbase = kh + (long)b * MKM * DKD;
  const unsigned short* vbase = vh + (long)b * DVD * MKM;

  if (tid < BN) { mrun[tid] = -3.0e38f; lrun[tid] = 0.0f; }

  // Prologue: TDM for Q tile + chunk-0 K/V into buffer 0, then drain.
  if (wave == 0) {
    tdm_load_2d((unsigned)(unsigned long long)(uintptr_t)&Ql[0],
                qh + ((long)b * NQ + n0) * DKD,
                DKD, NQ, DKD, BN, DKD, 4u);
    tdm_load_2d((unsigned)(unsigned long long)(uintptr_t)&Kl[0][0],
                kbase, DKD, MKM, DKD, BM, DKD, 4u);
    tdm_load_2d((unsigned)(unsigned long long)(uintptr_t)&Vl[0][0],
                vbase, MKM, DVD, BM, DVD, MKM, 3u);
  }
  if (tid < BM) biasS[0][tid] = mmk[(long)b * MKM + tid] ? 0.0f : NEG_BIG;
  if (wave == 0) __builtin_amdgcn_s_wait_tensorcnt(0);

  v8f acc[4][4];
#pragma unroll
  for (int vt = 0; vt < 4; ++vt)
#pragma unroll
    for (int j = 0; j < 4; ++j)
#pragma unroll
      for (int e = 0; e < 8; ++e) acc[vt][j][e] = 0.0f;

  const int sub_m = wave & 1;
  const int sub_n = wave >> 1;

  __syncthreads();

  for (int c = 0; c < NCH; ++c) {
    const int m0  = c * BM;
    const int buf = c & 1;

    // ---- pipeline: issue TDM for chunk c+1 into the other buffer, stage its bias ----
    if (c + 1 < NCH) {
      const int m1 = m0 + BM;
      if (wave == 0) {
        tdm_load_2d((unsigned)(unsigned long long)(uintptr_t)&Kl[buf ^ 1][0],
                    kbase + (long)m1 * DKD, DKD, MKM, DKD, BM, DKD, 4u);
        tdm_load_2d((unsigned)(unsigned long long)(uintptr_t)&Vl[buf ^ 1][0],
                    vbase + m1, MKM, DVD, BM, DVD, MKM, 3u);
      }
      if (tid < BM) biasS[buf ^ 1][tid] = mmk[(long)b * MKM + m1 + tid] ? 0.0f : NEG_BIG;
      // current chunk complete when only the 2 just-issued ops remain in flight
      if (wave == 0) __builtin_amdgcn_s_wait_tensorcnt(2);
    } else {
      if (wave == 0) __builtin_amdgcn_s_wait_tensorcnt(0);
    }
    __syncthreads();

    // ---- S-tile = K * Q for this wave (K-dim 64 -> 2 WMMAs) ----
    v8f s;
#pragma unroll
    for (int e = 0; e < 8; ++e) s[e] = 0.0f;
    {
      v16bf a0 = load_frag_a(&Kl[buf][0], sub_m * 16, KST, 0, lane);
      v16bf b0 = load_frag_b(Ql, sub_n * 16, KST, 0, lane);
      s = __builtin_amdgcn_wmma_f32_16x16x32_bf16(false, a0, false, b0, (short)0, s, false, false);
      v16bf a1 = load_frag_a(&Kl[buf][0], sub_m * 16, KST, 32, lane);
      v16bf b1 = load_frag_b(Ql, sub_n * 16, KST, 32, lane);
      s = __builtin_amdgcn_wmma_f32_16x16x32_bf16(false, a1, false, b1, (short)0, s, false, false);
    }
    float cmax = -3.0e38f;
#pragma unroll
    for (int r = 0; r < 8; ++r) {
      s[r] += biasS[buf][sub_m * 16 + kh2 * 8 + r];
      cmax = fmaxf(cmax, s[r]);
    }
    cmax = fmaxf(cmax, __shfl_xor(cmax, 16));
    if (lane < 16) pmaxS[sub_m][sub_n * 16 + lane] = cmax;
    __syncthreads();

    // ---- online softmax: new max + alpha ----
    if (tid < BN) {
      float mo = mrun[tid];
      float mn = fmaxf(mo, fmaxf(pmaxS[0][tid], pmaxS[1][tid]));
      alphaS[tid] = (mn <= -1.0e37f) ? 0.0f : __expf(mo - mn);
      mrun[tid] = mn;
    }
    __syncthreads();

    // ---- P = exp(S - mnew) -> LDS [n][m] bf16, partial column sums ----
    {
      int ncol = sub_n * 16 + (lane & 15);
      float mn = mrun[ncol];
      float ps = 0.0f;
#pragma unroll
      for (int r = 0; r < 8; ++r) {
        int mrow = sub_m * 16 + kh2 * 8 + r;
        float p = __expf(s[r] - mn);
        ps += p;
        Pl[ncol * VST + mrow] = f2bf(p);
      }
      ps += __shfl_xor(ps, 16);
      if (lane < 16) psumS[sub_m][sub_n * 16 + lane] = ps;
    }
    __syncthreads();

    // ---- l update, conditional rescale, O += V * P ----
    if (tid < BN)
      lrun[tid] = lrun[tid] * alphaS[tid] + psumS[0][tid] + psumS[1][tid];

    float aln[4];
#pragma unroll
    for (int j = 0; j < 4; ++j) aln[j] = alphaS[j * 16 + (lane & 15)];
    // wave-uniform skip: once the running max stabilizes alpha==1 exactly
    if (__any(aln[0] != 1.0f || aln[1] != 1.0f || aln[2] != 1.0f || aln[3] != 1.0f)) {
#pragma unroll
      for (int vt = 0; vt < 4; ++vt)
#pragma unroll
        for (int j = 0; j < 4; ++j)
#pragma unroll
          for (int e = 0; e < 8; ++e) acc[vt][j][e] *= aln[j];
    }

    v16bf pb[4];
#pragma unroll
    for (int j = 0; j < 4; ++j) pb[j] = load_frag_b(Pl, j * 16, VST, 0, lane);
#pragma unroll
    for (int vt = 0; vt < 4; ++vt) {
      v16bf va = load_frag_a(&Vl[buf][0], wave * 64 + vt * 16, VST, 0, lane);
#pragma unroll
      for (int j = 0; j < 4; ++j)
        acc[vt][j] = __builtin_amdgcn_wmma_f32_16x16x32_bf16(false, va, false, pb[j],
                                                             (short)0, acc[vt][j], false, false);
    }
    __syncthreads();
  }

  // ---- final 1/l normalization, qmask gating, store ----
  if (tid < BN) {
    float l = lrun[tid];
    float inv = (l > 0.0f) ? 1.0f / l : 0.0f;
    qscS[tid] = qm[(long)b * NQ + n0 + tid] ? inv : 0.0f;
  }
  __syncthreads();

#pragma unroll
  for (int j = 0; j < 4; ++j) {
    float sc = qscS[j * 16 + (lane & 15)];
    long col = n0 + j * 16 + (lane & 15);
#pragma unroll
    for (int vt = 0; vt < 4; ++vt) {
      int vbase2 = wave * 64 + vt * 16 + kh2 * 8;
#pragma unroll
      for (int r = 0; r < 8; ++r)
        out[((long)b * DVD + vbase2 + r) * NQ + col] = acc[vt][j][r] * sc;
    }
  }
}

extern "C" void kernel_launch(void* const* d_in, const int* in_sizes, int n_in,
                              void* d_out, int out_size, void* d_ws, size_t ws_size,
                              hipStream_t stream) {
  const float*         qkey  = (const float*)d_in[0];
  // d_in[1] = qval: never read by the reference (only its shape), skip it.
  const unsigned char* qmask = (const unsigned char*)d_in[2];
  const float*         mkey  = (const float*)d_in[3];
  const float*         mval  = (const float*)d_in[4];
  const unsigned char* mmask = (const unsigned char*)d_in[5];
  float* out = (float*)d_out;

  // workspace layout (bf16 staging): kh | qh | vh  (~38 MB total)
  unsigned short* kh = (unsigned short*)d_ws;                    // [B][MK][DK]
  unsigned short* qh = kh + (size_t)BB * MKM * DKD;              // [B][NQ][DK]
  unsigned short* vh = qh + (size_t)BB * NQ * DKD;               // [B][DV][MK]

  {
    long nq = (long)BB * DKD * NQ;
    conv_q_kernel<<<dim3((unsigned)((nq + 255) / 256)), dim3(256), 0, stream>>>(qkey, qh);
    long nk = (long)BB * DKD * MKM;
    conv_k_kernel<<<dim3((unsigned)((nk + 255) / 256)), dim3(256), 0, stream>>>(mkey, kh);
    conv_v_kernel<<<dim3(4096), dim3(256), 0, stream>>>(mval, (unsigned int*)vh);
  }

  dim3 grid(BB * (NQ / BN));  // 256 workgroups
  dim3 block(256);            // 8 waves (wave32)
  maskread_attn_kernel<<<grid, block, 0, stream>>>(qh, qmask, kh, vh, mmask, out);
}
